// FeatureExtractor_82128364634637
// MI455X (gfx1250) — matrix operand order repaired and checked by reading, not compiled, not run on hardware
//
#include <hip/hip_runtime.h>

typedef __attribute__((ext_vector_type(16))) _Float16 v16h;
typedef __attribute__((ext_vector_type(8)))  float    v8f;

namespace {
constexpr int S_C   = 4;
constexpr int B_C   = 64;
constexpr int NPG_C = 400;
constexpr int EMB_C = 64;
constexpr int G_C   = S_C * B_C;        // 256 graphs
constexpr int N_C   = G_C * NPG_C;      // 102400 nodes
constexpr int EPG_C = 6400;
constexpr int E_C   = G_C * EPG_C;      // 1,638,400 edges
constexpr int ETOT_C = E_C + N_C;       // + self loops = 1,740,800
constexpr float SLOPE_C = 0.2f;
}

// ---------------------------------------------------------------------------
// Weight prep: f32 [K,64] -> f16 B-fragments, pre-swizzled for WMMA SRC1.
// Fragment layout per (kt, nt): 32 lanes x 16 halves, element i of lane l is
// B[kt*32 + (l>>4)*16 + i][nt*16 + (l&15)]  (K striped over lane halves).
// ---------------------------------------------------------------------------
__global__ void prep_weight_kernel(const float* __restrict__ W, int K, int KT,
                                   _Float16* __restrict__ frag) {
  int t = blockIdx.x * blockDim.x + threadIdx.x;
  int total = KT * 4 * 32 * 16;
  if (t >= total) return;
  int i    = t & 15;
  int lane = (t >> 4) & 31;
  int nt   = (t >> 9) & 3;
  int kt   = t >> 11;
  int k   = kt * 32 + (lane >> 4) * 16 + i;
  int col = nt * 16 + (lane & 15);
  float v = (k < K) ? W[k * EMB_C + col] : 0.0f;
  frag[t] = (_Float16)v;
}

// ---------------------------------------------------------------------------
// Node transform: xl = h@Wl + bl, xr = h@Wr + br  via v_wmma_f32_16x16x32_f16.
// One wave per 16-node tile; 4 waves (128 threads) per block.
// A-fragment (16x32 f16, MxK): lane holds row l&15, element i is
// K half*16 + i (half = l>>4), matching the B fragment order.
// C/D (16x16 f32): lane col = l&15, VGPR i -> row = 8*(l>>4) + i.
// ---------------------------------------------------------------------------
template <int K, int KT>
__global__ void __launch_bounds__(128)
transform_kernel(const float* __restrict__ h, int ldh,
                 const _Float16* __restrict__ fragWl,
                 const _Float16* __restrict__ fragWr,
                 const float* __restrict__ bl, const float* __restrict__ br,
                 float* __restrict__ xl, float* __restrict__ xr) {
  const int lane = threadIdx.x & 31;
  const int wave = threadIdx.x >> 5;
  const int tile = blockIdx.x * 4 + wave;   // 16-node tile index
  const int half = lane >> 4;
  const int colc = lane & 15;
  const int row  = tile * 16 + colc;        // A row for this lane

  // Build A fragments (KT K-steps of 32)
  v16h a[KT];
#pragma unroll
  for (int kt = 0; kt < KT; ++kt) {
#pragma unroll
    for (int i = 0; i < 16; ++i) {
      int kg = kt * 32 + half * 16 + i;
      float v = (kg < K) ? h[row * ldh + kg] : 0.0f;
      a[kt][i] = (_Float16)v;
    }
  }

#pragma unroll
  for (int sel = 0; sel < 2; ++sel) {
    const _Float16* frag = sel ? fragWr : fragWl;
    const float* bias    = sel ? br : bl;
    float* dst           = sel ? xr : xl;
#pragma unroll
    for (int nt = 0; nt < 4; ++nt) {
      float bv = bias[nt * 16 + colc];
      v8f c;
#pragma unroll
      for (int i = 0; i < 8; ++i) c[i] = bv;   // accumulate on top of bias
#pragma unroll
      for (int kt = 0; kt < KT; ++kt) {
        v16h b = *(const v16h*)(frag + ((size_t)(kt * 4 + nt) * 32 + lane) * 16);
        c = __builtin_amdgcn_wmma_f32_16x16x32_f16(
            false, a[kt], false, b, (short)0, c, false, false);
      }
#pragma unroll
      for (int i = 0; i < 8; ++i)
        dst[(size_t)(tile * 16 + half * 8 + i) * EMB_C + nt * 16 + colc] = c[i];
    }
  }
}

// ---------------------------------------------------------------------------
// Per-layer scratch init
// ---------------------------------------------------------------------------
__global__ void init_nodes_kernel(float* __restrict__ mx, float* __restrict__ den) {
  int n = blockIdx.x * blockDim.x + threadIdx.x;
  if (n < N_C) { mx[n] = -INFINITY; den[n] = 0.0f; }
}

__global__ void init_out_kernel(float* __restrict__ outb, const float* __restrict__ bias) {
  int t = blockIdx.x * blockDim.x + threadIdx.x;
  if (t < N_C * EMB_C) outb[t] = bias[t & (EMB_C - 1)];
}

// Sign-aware float atomic max (maps to global_atomic_max_i32 / min_u32)
__device__ __forceinline__ void atomicMaxF(float* addr, float val) {
  if (val >= 0.0f)
    atomicMax((int*)addr, __float_as_int(val));
  else
    atomicMin((unsigned int*)addr, __float_as_uint(val));
}

// ---------------------------------------------------------------------------
// Edge pass 1: e = att . leaky_relu(xl[src] + xr[dst]); segment max over dst
// ---------------------------------------------------------------------------
__global__ void edge_score_kernel(const long long* __restrict__ esrc,
                                  const long long* __restrict__ edst,
                                  const float* __restrict__ xl,
                                  const float* __restrict__ xr,
                                  const float* __restrict__ att,
                                  float* __restrict__ esc,
                                  float* __restrict__ mx) {
  int e = blockIdx.x * blockDim.x + threadIdx.x;
  if (e >= ETOT_C) return;
  int s, d;
  if (e < E_C) { s = (int)esrc[e]; d = (int)edst[e]; }
  else         { s = e - E_C; d = s; }
  const float4* pl = (const float4*)(xl + (size_t)s * EMB_C);
  const float4* pr = (const float4*)(xr + (size_t)d * EMB_C);
  const float4* pa = (const float4*)att;
  float acc = 0.0f;
#pragma unroll
  for (int j = 0; j < 16; ++j) {
    float4 a = pl[j], b = pr[j], w = pa[j];
    float v;
    v = a.x + b.x; v = (v > 0.0f) ? v : SLOPE_C * v; acc += w.x * v;
    v = a.y + b.y; v = (v > 0.0f) ? v : SLOPE_C * v; acc += w.y * v;
    v = a.z + b.z; v = (v > 0.0f) ? v : SLOPE_C * v; acc += w.z * v;
    v = a.w + b.w; v = (v > 0.0f) ? v : SLOPE_C * v; acc += w.w * v;
  }
  esc[e] = acc;
  atomicMaxF(&mx[d], acc);
}

// ---------------------------------------------------------------------------
// Edge pass 2: ex = exp(e - m[dst]); segment sum over dst
// ---------------------------------------------------------------------------
__global__ void edge_exp_kernel(const long long* __restrict__ edst,
                                float* __restrict__ esc,
                                const float* __restrict__ mx,
                                float* __restrict__ den) {
  int e = blockIdx.x * blockDim.x + threadIdx.x;
  if (e >= ETOT_C) return;
  int d = (e < E_C) ? (int)edst[e] : (e - E_C);
  float ex = __expf(esc[e] - mx[d]);
  esc[e] = ex;
  atomicAdd(&den[d], ex);
}

// ---------------------------------------------------------------------------
// Edge pass 3: out[dst] += (ex/den[dst]) * xl[src]; 16 threads/edge, float4
// ---------------------------------------------------------------------------
__global__ void edge_aggr_kernel(const long long* __restrict__ esrc,
                                 const long long* __restrict__ edst,
                                 const float* __restrict__ esc,
                                 const float* __restrict__ den,
                                 const float* __restrict__ xl,
                                 float* __restrict__ outb) {
  int t = blockIdx.x * blockDim.x + threadIdx.x;
  if (t >= ETOT_C * 16) return;
  int e = t >> 4;
  int q = t & 15;
  int s, d;
  if (e < E_C) { s = (int)esrc[e]; d = (int)edst[e]; }
  else         { s = e - E_C; d = s; }
  float alpha = esc[e] / den[d];
  float4 v = ((const float4*)(xl + (size_t)s * EMB_C))[q];
  float* o = outb + (size_t)d * EMB_C + q * 4;
  atomicAdd(o + 0, alpha * v.x);
  atomicAdd(o + 1, alpha * v.y);
  atomicAdd(o + 2, alpha * v.z);
  atomicAdd(o + 3, alpha * v.w);
}

__global__ void relu_kernel(const float* __restrict__ src, float* __restrict__ dst) {
  int t = blockIdx.x * blockDim.x + threadIdx.x;
  if (t < N_C * EMB_C) { float v = src[t]; dst[t] = (v > 0.0f) ? v : 0.0f; }
}

// ---------------------------------------------------------------------------
// Final assembly: out[n*67 + 0:64] = h;  [64]=dec, [65]=reachable, [66]=nn
// ---------------------------------------------------------------------------
__global__ void out_h_kernel(const float* __restrict__ hfin, float* __restrict__ out) {
  int t = blockIdx.x * blockDim.x + threadIdx.x;
  if (t >= N_C * EMB_C) return;
  int n = t >> 6, c = t & 63;
  out[(size_t)n * 67 + c] = hfin[t];
}

__global__ void out_meta_kernel(const int* __restrict__ reach, float* __restrict__ out) {
  int n = blockIdx.x * blockDim.x + threadIdx.x;
  if (n >= N_C) return;
  int j = n % (N_C / S_C);                       // index within S-chunk
  float* o = out + (size_t)n * 67;
  o[64] = (float)(j / NPG_C);                    // decoupled batch id
  o[65] = (float)reach[n];                       // reachable flag
  o[66] = (j < G_C) ? (float)NPG_C : 0.0f;       // num_nodes column
}

// ---------------------------------------------------------------------------
extern "C" void kernel_launch(void* const* d_in, const int* in_sizes, int n_in,
                              void* d_out, int out_size, void* d_ws, size_t ws_size,
                              hipStream_t stream) {
  const float*     x     = (const float*)d_in[0];
  const long long* ei    = (const long long*)d_in[1];
  const int*       reach = (const int*)d_in[2];
  const float* Wl0  = (const float*)d_in[3];
  const float* Wr0  = (const float*)d_in[4];
  const float* bl0  = (const float*)d_in[5];
  const float* br0  = (const float*)d_in[6];
  const float* att0 = (const float*)d_in[7];
  const float* bias0= (const float*)d_in[8];
  const float* Wl   = (const float*)d_in[9];    // [4,64,64]
  const float* Wr   = (const float*)d_in[10];
  const float* bl   = (const float*)d_in[11];   // [4,64]
  const float* br   = (const float*)d_in[12];
  const float* att  = (const float*)d_in[13];
  const float* bias = (const float*)d_in[14];
  float* out = (float*)d_out;

  // workspace carve-up (~113 MB)
  float* ws   = (float*)d_ws;
  float* h    = ws;
  float* xl   = h    + (size_t)N_C * EMB_C;
  float* xr   = xl   + (size_t)N_C * EMB_C;
  float* outb = xr   + (size_t)N_C * EMB_C;
  float* esc  = outb + (size_t)N_C * EMB_C;
  float* mx   = esc  + ETOT_C;
  float* den  = mx   + N_C;
  _Float16* frag = (_Float16*)(den + N_C);
  _Float16* f0l  = frag;                  // layer0 Wl frag (KT=1): 2048 halves
  _Float16* f0r  = frag + 2048;
  _Float16* fl1  = frag + 4096;           // 4 x 4096 halves
  _Float16* fr1  = frag + 4096 + 4 * 4096;

  const long long* esrc = ei;
  const long long* edst = ei + E_C;

  const int TB = 256;
  const int gN    = (N_C + TB - 1) / TB;                 // 400
  const int gNE   = (N_C * EMB_C + TB - 1) / TB;         // 25600
  const int gE    = (ETOT_C + TB - 1) / TB;              // 6800
  const int gE16  = (ETOT_C * 16 + TB - 1) / TB;         // 108800
  const int gXform = (N_C / 16) / 4;                     // 1600 blocks of 128

  // --- weight prep (tiny, cached in L2 thereafter) ---
  prep_weight_kernel<<<8, TB, 0, stream>>>(Wl0, 7, 1, f0l);
  prep_weight_kernel<<<8, TB, 0, stream>>>(Wr0, 7, 1, f0r);
  for (int i = 0; i < 4; ++i) {
    prep_weight_kernel<<<16, TB, 0, stream>>>(Wl + (size_t)i * EMB_C * EMB_C, 64, 2, fl1 + (size_t)i * 4096);
    prep_weight_kernel<<<16, TB, 0, stream>>>(Wr + (size_t)i * EMB_C * EMB_C, 64, 2, fr1 + (size_t)i * 4096);
  }

  auto run_edges = [&](const float* attp, const float* biasp) {
    init_nodes_kernel<<<gN, TB, 0, stream>>>(mx, den);
    init_out_kernel<<<gNE, TB, 0, stream>>>(outb, biasp);
    edge_score_kernel<<<gE, TB, 0, stream>>>(esrc, edst, xl, xr, attp, esc, mx);
    edge_exp_kernel<<<gE, TB, 0, stream>>>(edst, esc, mx, den);
    edge_aggr_kernel<<<gE16, TB, 0, stream>>>(esrc, edst, esc, den, xl, outb);
  };

  // --- layer 0: x[N,7] -> h[N,64], relu ---
  transform_kernel<7, 1><<<gXform, 128, 0, stream>>>(x, 7, f0l, f0r, bl0, br0, xl, xr);
  run_edges(att0, bias0);
  relu_kernel<<<gNE, TB, 0, stream>>>(outb, h);

  // --- layers 1..4 ---
  for (int i = 0; i < 4; ++i) {
    transform_kernel<64, 2><<<gXform, 128, 0, stream>>>(
        h, EMB_C, fl1 + (size_t)i * 4096, fr1 + (size_t)i * 4096,
        bl + i * EMB_C, br + i * EMB_C, xl, xr);
    run_edges(att + i * EMB_C, bias + i * EMB_C);
    if (i < 3) relu_kernel<<<gNE, TB, 0, stream>>>(outb, h);
  }

  // --- final packing (outb holds last layer, no relu) ---
  out_h_kernel<<<gNE, TB, 0, stream>>>(outb, out);
  out_meta_kernel<<<gN, TB, 0, stream>>>(reach, out);
}
